// BidirectionalMambaBlock_73263552135383
// MI455X (gfx1250) — compile-verified, matrix operand order checked by reading, not run on hardware
//
#include <hip/hip_runtime.h>
#include <math.h>

// ---------------------------------------------------------------------------
// BidirectionalMambaBlock on gfx1250 (MI455X).
// B=8, L=2048, D=512, D_INNER=1024, D_STATE=16, DT_RANK=32, D_CONV=4, HIDDEN=1024
//
// Projections: f16 operands (converted once), fp32-accumulating
// v_wmma_f32_16x16x32_f16. Tiles move via gfx1250 async DMA
// (global_load_async_to_lds_b128) in a double-buffered pipeline overlapped
// with WMMA using partial s_wait_asynccnt waits.
// Selective scan: fp32 recurrence in registers, one thread per (b, d).
// ---------------------------------------------------------------------------

#define BSZ      8
#define LSEQ     2048
#define LMASK    (LSEQ - 1)
#define DMODEL   512
#define DINNER   1024
#define DSTATE   16
#define DTRANK   32
#define DCONV    4
#define HIDDEN_  1024
#define ROWS     (BSZ * LSEQ)          // 16384

#define TILE_M   128
#define TILE_N   64
#define TILE_K   32
#define LDSTR    40                    // f16 elems/row; 80 B = multiple of 16 B

typedef __attribute__((ext_vector_type(16))) _Float16 v16h;
typedef __attribute__((ext_vector_type(8)))  _Float16 v8h;
typedef __attribute__((ext_vector_type(8)))  float    v8f;

enum { ACT_NONE = 0, ACT_RELU = 1, ACT_SOFTPLUS = 2 };

// ---- gfx1250 async global->LDS DMA (16 B per lane), ASYNCcnt-tracked -------
__device__ __forceinline__ void async_copy_b128(const _Float16* gsrc, _Float16* lds_dst)
{
    // Low 32 bits of a generic LDS pointer are the LDS byte address.
    unsigned lds_addr = (unsigned)(unsigned long long)lds_dst;
    asm volatile("global_load_async_to_lds_b128 %0, %1, off"
                 :
                 : "v"(lds_addr), "v"(gsrc)
                 : "memory");
}
__device__ __forceinline__ void async_wait_le3()
{
    asm volatile("s_wait_asynccnt 0x3" ::: "memory");   // prev tile done, next in flight
}
__device__ __forceinline__ void async_wait_le0()
{
    asm volatile("s_wait_asynccnt 0x0" ::: "memory");
}

// ---------------------------------------------------------------------------
// WMMA GEMM: C[M,N] = act(A[M,K] @ W[N,K]^T + bias), A/W in f16, C in OutT.
// Block 128x64, 256 threads = 8 waves; each wave owns a 16x64 strip
// (1 A-fragment -> 4 WMMAs per 32-deep K step). Double-buffered async DMA.
// rev_in / rev_out reverse the L index inside each batch block (row ^ LMASK).
// ---------------------------------------------------------------------------
template <int ACT, typename OutT>
__global__ __launch_bounds__(256)
void gemm_wmma_kernel(const _Float16* __restrict__ A, const _Float16* __restrict__ W,
                      const float* __restrict__ bias, OutT* __restrict__ C,
                      int M, int N, int K, int lda, int ldc,
                      int rev_in, int rev_out)
{
    __shared__ alignas(16) _Float16 As[2][TILE_M][LDSTR];
    __shared__ alignas(16) _Float16 Bs[2][TILE_N][LDSTR];

    const int tid  = threadIdx.x;
    const int wave = tid >> 5;
    const int lane = tid & 31;
    const int hf   = lane >> 4;        // K-octet group of this lane
    const int l16  = lane & 15;

    const int m0 = blockIdx.y * TILE_M;
    const int n0 = blockIdx.x * TILE_N;

    // Issue one tile's async copies: 3 async instructions per wave.
    auto issue_tile = [&](int buf, int k0) {
        // A tile: 128 rows x 4 x 16B segments = 512 lane-transfers (2 instr)
#pragma unroll
        for (int j = 0; j < 2; ++j) {
            const int id   = tid + j * 256;
            const int rowl = id >> 2;
            const int seg  = id & 3;
            unsigned grow = (unsigned)(m0 + rowl);
            if (rev_in) grow ^= LMASK;           // reverse l within batch block
            async_copy_b128(A + (long)grow * lda + k0 + seg * 8,
                            &As[buf][rowl][seg * 8]);
        }
        // B tile: 64 rows x 4 segments = 256 lane-transfers (1 instr)
        const int rowl = tid >> 2;
        const int seg  = tid & 3;
        async_copy_b128(W + (long)(n0 + rowl) * K + k0 + seg * 8,
                        &Bs[buf][rowl][seg * 8]);
    };

    v8f acc[4] = {{}, {}, {}, {}};

    const int nk = K / TILE_K;
    issue_tile(0, 0);

    for (int kt = 0; kt < nk; ++kt) {
        const int cur = kt & 1;
        if (kt + 1 < nk) {
            issue_tile(cur ^ 1, (kt + 1) * TILE_K);
            async_wait_le3();                    // tile kt complete; kt+1 in flight
        } else {
            async_wait_le0();
        }
        __syncthreads();

        // Fragments per CDNA5 16-bit A/B VGPR layout:
        // lane (l16, hf): row = l16, K = {koff..koff+7, koff+16..koff+23}, koff = hf*8
        const int mrow = wave * 16 + l16;
        const int koff = hf * 8;
        union F16x16 { v16h v; v8h h[2]; };
        F16x16 af, bf[4];
        af.h[0] = *(const v8h*)&As[cur][mrow][koff];
        af.h[1] = *(const v8h*)&As[cur][mrow][koff + 16];
#pragma unroll
        for (int t = 0; t < 4; ++t) {
            bf[t].h[0] = *(const v8h*)&Bs[cur][t * 16 + l16][koff];
            bf[t].h[1] = *(const v8h*)&Bs[cur][t * 16 + l16][koff + 16];
        }
#pragma unroll
        for (int t = 0; t < 4; ++t) {
            acc[t] = __builtin_amdgcn_wmma_f32_16x16x32_f16(false, af.v, false, bf[t].v,
                                                            (short)0, acc[t], false, false);
        }
        __syncthreads();                         // buffer reuse fence
    }

    // ---- epilogue: C VGPR i -> M = i + 8*hf, N = l16 (+16 per sub-tile) ----
#pragma unroll
    for (int i = 0; i < 8; ++i) {
        unsigned grow = (unsigned)(m0 + wave * 16 + hf * 8 + i);
        if (rev_out) grow ^= LMASK;
        OutT* dst = C + (long)grow * ldc + n0;
#pragma unroll
        for (int t = 0; t < 4; ++t) {
            const int c = t * 16 + l16;
            float v = acc[t][i];
            if (bias) v += bias[n0 + c];
            if (ACT == ACT_RELU) {
                v = v > 0.f ? v : 0.f;
            } else if (ACT == ACT_SOFTPLUS) {
                v = (v > 20.f) ? v : log1pf(__expf(v));
            }
            dst[c] = (OutT)v;
        }
    }
}

// ---------------------------------------------------------------------------
__global__ __launch_bounds__(256)
void cvt_f32_to_f16_kernel(const float* __restrict__ src, _Float16* __restrict__ dst,
                           long n)
{
    long i = (long)blockIdx.x * blockDim.x + threadIdx.x;
    const long stride = (long)gridDim.x * blockDim.x;
    for (; i < n; i += stride) dst[i] = (_Float16)src[i];
}

// ---------------------------------------------------------------------------
// Depthwise causal conv (k=4) + bias + SiLU on the xi half of xz (f16 in).
// Writes xi in f32 (for the scan) and f16 (for the x_proj GEMM).
// ---------------------------------------------------------------------------
__global__ __launch_bounds__(256)
void conv_silu_kernel(const _Float16* __restrict__ xz, const float* __restrict__ cw,
                      const float* __restrict__ cb, float* __restrict__ xi_f32,
                      _Float16* __restrict__ xi_f16)
{
    const long idx = (long)blockIdx.x * blockDim.x + threadIdx.x;
    const int d = (int)(idx & (DINNER - 1));
    const long r = idx >> 10;                     // row = b*L + l
    const int l = (int)(r & LMASK);

    float acc = cb[d];
#pragma unroll
    for (int j = 0; j < DCONV; ++j) {
        const int ll = l - (DCONV - 1) + j;
        if (ll >= 0) {
            acc += cw[d * DCONV + j] *
                   (float)xz[(r + (long)(ll - l)) * (2 * DINNER) + d];
        }
    }
    acc = acc / (1.f + __expf(-acc));             // SiLU
    xi_f32[r * DINNER + d] = acc;
    xi_f16[r * DINNER + d] = (_Float16)acc;
}

// ---------------------------------------------------------------------------
// Selective scan: one thread per (b, d); 16-wide fp32 state in registers.
// ---------------------------------------------------------------------------
__global__ __launch_bounds__(256)
void scan_kernel(const float* __restrict__ xi, const float* __restrict__ dt,
                 const _Float16* __restrict__ xdbl16, const _Float16* __restrict__ xz16,
                 const float* __restrict__ A_log, const float* __restrict__ Dp,
                 _Float16* __restrict__ ygate16)
{
    const int gid = blockIdx.x * blockDim.x + threadIdx.x;   // 0..8191
    const int d = gid & (DINNER - 1);
    const int b = gid >> 10;

    float a[DSTATE], h[DSTATE];
#pragma unroll
    for (int s = 0; s < DSTATE; ++s) {
        a[s] = -__expf(A_log[d * DSTATE + s]);
        h[s] = 0.f;
    }
    const float Dv = Dp[d];

    for (int l = 0; l < LSEQ; ++l) {
        const long r = (long)b * LSEQ + l;
        const float dtv = dt[r * DINNER + d];
        const float xv  = xi[r * DINNER + d];
        const float dx  = dtv * xv;
        const _Float16* bc = xdbl16 + r * (DTRANK + 2 * DSTATE);
        if (l + 1 < LSEQ) {
            __builtin_prefetch(bc + (DTRANK + 2 * DSTATE), 0, 1);
        }
        float y = 0.f;
#pragma unroll
        for (int s = 0; s < DSTATE; ++s) {
            const float dA = __expf(dtv * a[s]);
            h[s] = dA * h[s] + dx * (float)bc[DTRANK + s];
            y += h[s] * (float)bc[DTRANK + DSTATE + s];
        }
        const float zv = (float)xz16[r * (2 * DINNER) + DINNER + d];
        y = (y + xv * Dv) * (zv / (1.f + __expf(-zv)));
        ygate16[r * DINNER + d] = (_Float16)y;
    }
}

// ---------------------------------------------------------------------------
// out = LayerNorm(a + b [+ c]) * w + bias over D=512; optional f16 mirror.
// ---------------------------------------------------------------------------
__global__ __launch_bounds__(256)
void add_ln_kernel(const float* __restrict__ a, const float* __restrict__ b,
                   const float* __restrict__ c, const float* __restrict__ w,
                   const float* __restrict__ bias, float* __restrict__ out,
                   _Float16* __restrict__ outh)
{
    const int t = threadIdx.x;
    const long base = (long)blockIdx.x * DMODEL;
    const int i0 = t, i1 = t + 256;

    float v0 = a[base + i0] + b[base + i0] + (c ? c[base + i0] : 0.f);
    float v1 = a[base + i1] + b[base + i1] + (c ? c[base + i1] : 0.f);

    __shared__ float s1[256];
    __shared__ float s2[256];
    s1[t] = v0 + v1;
    s2[t] = v0 * v0 + v1 * v1;
    __syncthreads();
    for (int off = 128; off > 0; off >>= 1) {
        if (t < off) { s1[t] += s1[t + off]; s2[t] += s2[t + off]; }
        __syncthreads();
    }
    const float mean = s1[0] * (1.f / DMODEL);
    const float var  = s2[0] * (1.f / DMODEL) - mean * mean;
    const float rstd = rsqrtf(var + 1e-5f);
    const float o0 = (v0 - mean) * rstd * w[i0] + bias[i0];
    const float o1 = (v1 - mean) * rstd * w[i1] + bias[i1];
    out[base + i0] = o0;
    out[base + i1] = o1;
    if (outh) {
        outh[base + i0] = (_Float16)o0;
        outh[base + i1] = (_Float16)o1;
    }
}

// ---------------------------------------------------------------------------
extern "C" void kernel_launch(void* const* d_in, const int* in_sizes, int n_in,
                              void* d_out, int out_size, void* d_ws, size_t ws_size,
                              hipStream_t stream)
{
    (void)in_sizes; (void)n_in; (void)out_size; (void)ws_size;

    const float* x    = (const float*)d_in[0];
    const float* pu_w = (const float*)d_in[19];
    const float* pu_b = (const float*)d_in[20];
    const float* pl_w = (const float*)d_in[21];
    const float* pl_b = (const float*)d_in[22];
    const float* ln_w = (const float*)d_in[23];
    const float* ln_b = (const float*)d_in[24];

    const long R = (long)ROWS;

    // ---- carve workspace (256 B aligned chunks) ----
    char* wp = (char*)d_ws;
    auto carve = [&](size_t bytes) -> void* {
        void* q = (void*)wp;
        wp += (bytes + 255) & ~(size_t)255;
        return q;
    };
    _Float16* x16    = (_Float16*)carve(R * DMODEL * 2);           // 16.8 MB
    _Float16* xz16   = (_Float16*)carve(R * 2 * DINNER * 2);       // 67 MB (reused: MLP hidden)
    float*    xi32   = (float*)   carve(R * DINNER * 4);           // 67 MB
    _Float16* xi16   = (_Float16*)carve(R * DINNER * 2);           // 33.5 MB (reused: gated y)
    _Float16* xdbl16 = (_Float16*)carve(R * (DTRANK + 2 * DSTATE) * 2);
    float*    dtb    = (float*)   carve(R * DINNER * 4);           // 67 MB
    float*    y1     = (float*)   carve(R * DMODEL * 4);
    float*    y2     = (float*)   carve(R * DMODEL * 4);
    float*    y3     = (float*)   carve(R * DMODEL * 4);
    float*    yp     = (float*)   carve(R * DMODEL * 4);
    _Float16* y3h    = (_Float16*)carve(R * DMODEL * 2);
    _Float16* inw16[2], *xpj16[2], *dtw16[2], *outw16[2];
    for (int p = 0; p < 2; ++p) {
        inw16[p]  = (_Float16*)carve((size_t)2 * DINNER * DMODEL * 2);
        xpj16[p]  = (_Float16*)carve((size_t)(DTRANK + 2 * DSTATE) * DINNER * 2);
        dtw16[p]  = (_Float16*)carve((size_t)DINNER * DTRANK * 2);
        outw16[p] = (_Float16*)carve((size_t)DMODEL * DINNER * 2);
    }
    _Float16* pu16 = (_Float16*)carve((size_t)HIDDEN_ * DMODEL * 2);
    _Float16* pl16 = (_Float16*)carve((size_t)DMODEL * HIDDEN_ * 2);

    const dim3 blk(256);
    const dim3 cvtg(2048);
    auto ggrid = [](int N) { return dim3(N / TILE_N, ROWS / TILE_M); };

    // ---- one-time f32 -> f16 operand conversion ----
    cvt_f32_to_f16_kernel<<<cvtg, blk, 0, stream>>>(x, x16, R * DMODEL);
    for (int p = 0; p < 2; ++p) {
        const int base = 1 + p * 9;
        cvt_f32_to_f16_kernel<<<cvtg, blk, 0, stream>>>((const float*)d_in[base + 0], inw16[p],  (long)2 * DINNER * DMODEL);
        cvt_f32_to_f16_kernel<<<cvtg, blk, 0, stream>>>((const float*)d_in[base + 3], xpj16[p],  (long)(DTRANK + 2 * DSTATE) * DINNER);
        cvt_f32_to_f16_kernel<<<cvtg, blk, 0, stream>>>((const float*)d_in[base + 4], dtw16[p],  (long)DINNER * DTRANK);
        cvt_f32_to_f16_kernel<<<cvtg, blk, 0, stream>>>((const float*)d_in[base + 8], outw16[p], (long)DMODEL * DINNER);
    }
    cvt_f32_to_f16_kernel<<<cvtg, blk, 0, stream>>>(pu_w, pu16, (long)HIDDEN_ * DMODEL);
    cvt_f32_to_f16_kernel<<<cvtg, blk, 0, stream>>>(pl_w, pl16, (long)DMODEL * HIDDEN_);

    // ---- two mamba passes (p=1 runs on the L-reversed sequence) ----
    for (int p = 0; p < 2; ++p) {
        const int base = 1 + p * 9;
        const float* conv_w = (const float*)d_in[base + 1];
        const float* conv_b = (const float*)d_in[base + 2];
        const float* dt_b   = (const float*)d_in[base + 5];
        const float* A_log  = (const float*)d_in[base + 6];
        const float* Dp     = (const float*)d_in[base + 7];
        const int rev = p;

        // xz = x(_rev) @ in_w^T                       [R, 2048] f16
        gemm_wmma_kernel<ACT_NONE, _Float16><<<ggrid(2 * DINNER), blk, 0, stream>>>(
            x16, inw16[p], nullptr, xz16, ROWS, 2 * DINNER, DMODEL, DMODEL, 2 * DINNER, rev, 0);

        // xi = silu(causal_dwconv(xz[:, :1024]))      [R, 1024] f32 + f16
        conv_silu_kernel<<<dim3((unsigned)(R * DINNER / 256)), blk, 0, stream>>>(
            xz16, conv_w, conv_b, xi32, xi16);

        // xdbl = xi @ xproj_w^T                       [R, 64] f16
        gemm_wmma_kernel<ACT_NONE, _Float16><<<ggrid(DTRANK + 2 * DSTATE), blk, 0, stream>>>(
            xi16, xpj16[p], nullptr, xdbl16, ROWS, DTRANK + 2 * DSTATE, DINNER,
            DINNER, DTRANK + 2 * DSTATE, 0, 0);

        // dt = softplus(xdbl[:, :32] @ dt_w^T + dt_b) [R, 1024] f32
        gemm_wmma_kernel<ACT_SOFTPLUS, float><<<ggrid(DINNER), blk, 0, stream>>>(
            xdbl16, dtw16[p], dt_b, dtb, ROWS, DINNER, DTRANK,
            DTRANK + 2 * DSTATE, DINNER, 0, 0);

        // selective scan + D skip + SiLU(z) gate -> gated y (f16, reuses xi16)
        scan_kernel<<<dim3(BSZ * DINNER / 256), blk, 0, stream>>>(
            xi32, dtb, xdbl16, xz16, A_log, Dp, xi16);

        // y = gated @ out_w^T  (rev_out un-reverses the m2 pass)   [R, 512] f32
        gemm_wmma_kernel<ACT_NONE, float><<<ggrid(DMODEL), blk, 0, stream>>>(
            xi16, outw16[p], nullptr, (p == 0 ? y1 : y2), ROWS, DMODEL, DINNER,
            DINNER, DMODEL, 0, rev);
    }

    // y3 = LN(x + y1 + y2)  (+ f16 mirror for the MLP)
    add_ln_kernel<<<dim3(ROWS), blk, 0, stream>>>(x, y1, y2, ln_w, ln_b, y3, y3h);

    // hidden = relu(y3 @ pu_w^T + pu_b)   [R, 1024] f16 (reuses xz16)
    _Float16* hidden16 = xz16;
    gemm_wmma_kernel<ACT_RELU, _Float16><<<ggrid(HIDDEN_), blk, 0, stream>>>(
        y3h, pu16, pu_b, hidden16, ROWS, HIDDEN_, DMODEL, DMODEL, HIDDEN_, 0, 0);

    // yp = hidden @ pl_w^T + pl_b         [R, 512] f32
    gemm_wmma_kernel<ACT_NONE, float><<<ggrid(DMODEL), blk, 0, stream>>>(
        hidden16, pl16, pl_b, yp, ROWS, DMODEL, HIDDEN_, HIDDEN_, DMODEL, 0, 0);

    // out = LN(yp + y3)
    add_ln_kernel<<<dim3(ROWS), blk, 0, stream>>>(yp, y3, nullptr, ln_w, ln_b,
                                                  (float*)d_out, nullptr);
}
